// Mamba2_77412490543279
// MI455X (gfx1250) — matrix operand order repaired and checked by reading, not compiled, not run on hardware
//
#include <hip/hip_runtime.h>
#include <hip/hip_bf16.h>
#include <math.h>
#include <stdint.h>

typedef __attribute__((ext_vector_type(16))) __bf16 v16bf;
typedef __attribute__((ext_vector_type(8)))  __bf16 v8bf;
typedef __attribute__((ext_vector_type(8)))  float  v8f;
typedef __attribute__((ext_vector_type(4)))  unsigned int u32x4;
typedef __attribute__((ext_vector_type(8)))  int i32x8;
typedef __attribute__((ext_vector_type(4)))  int i32x4;

#define D_MODEL   1024
#define D_INNER   2048
#define N_HEADS   32
#define HEADDIM   64
#define D_CONV    4
#define SEQ_L     4096
#define BATCH     2
#define NROWS     (BATCH * SEQ_L)            // 8192 tokens
#define E_OUT     (2 * D_INNER + N_HEADS)    // 4128
#define EPSV      1e-5f
#define NCHUNK    32
#define CHUNKLEN  (SEQ_L / NCHUNK)           // 128
#define NCHAIN    (BATCH * N_HEADS * HEADDIM)// 4096
#define KC        512                         // K-chunk staged in LDS (32KB/buf)

#if defined(__HIP_DEVICE_COMPILE__) && __has_builtin(__builtin_amdgcn_tensor_load_to_lds)
#define HAVE_TDM 1
#else
#define HAVE_TDM 0
#endif

// ---------------------------------------------------------------------------
// f32 -> bf16 conversion
// ---------------------------------------------------------------------------
__global__ void cvt_f32_bf16(const float* __restrict__ in,
                             __bf16* __restrict__ out, long n) {
  long i = (long)blockIdx.x * blockDim.x + threadIdx.x;
  if (i < n) out[i] = (__bf16)in[i];
}

#if HAVE_TDM
// Issue one TDM load of a 32-row x KC-element bf16 strip (row stride K elems)
// from global into LDS at byte offset ldsByteOff.
__device__ __forceinline__ void tdm_load_strip(const __bf16* gsrc,
                                               unsigned ldsByteOff, int K) {
  const unsigned long long ga = (unsigned long long)(uintptr_t)gsrc;
  u32x4 g0;
  g0.x = 1u;                                   // count=1, user mode
  g0.y = ldsByteOff;                           // lds_addr (bytes)
  g0.z = (unsigned)(ga & 0xFFFFFFFFu);         // global_addr[31:0]
  g0.w = (unsigned)((ga >> 32) & 0x01FFFFFFu)  // global_addr[56:32]
         | (2u << 30);                         // type = 2 (image)
  i32x8 g1;
  g1[0] = 0x00010000;                          // data_size = 1 (2 bytes)
  g1[1] = (int)(((unsigned)KC & 0xFFFFu) << 16);   // tensor_dim0[15:0] = KC
  g1[2] = (int)(32u << 16);                    // tensor_dim1[15:0] = 32
  g1[3] = (int)(((unsigned)KC & 0xFFFFu) << 16);   // tile_dim0 = KC
  g1[4] = 32;                                  // tile_dim1 = 32
  g1[5] = K;                                   // tensor_dim0_stride = K
  g1[6] = 0;
  g1[7] = 0;
  i32x4 z4 = {0, 0, 0, 0};
#if defined(__clang_major__) && __clang_major__ >= 23
  i32x8 z8 = {0, 0, 0, 0, 0, 0, 0, 0};
  __builtin_amdgcn_tensor_load_to_lds(g0, g1, z4, z4, z8, 0);
#else
  __builtin_amdgcn_tensor_load_to_lds(g0, g1, z4, z4, 0);
#endif
}
#endif

// ---------------------------------------------------------------------------
// bf16 WMMA GEMM: C[M,N] = A[M,K] * W[N,K]^T  (fp32 accumulate)
// Block = 8 waves, 256x32 C tile. The 32-row weight strip is staged into LDS
// by the Tensor Data Mover, double-buffered: the DMA for chunk c+1 overlaps
// the 64 WMMAs of chunk c (TDM completes in order -> wait tensorcnt<=1).
//   A frag (16x32): lane m=lane%16, g=lane/16 -> K in {16g..16g+15}
//   B frag (32x16): lane n=lane%16, g=lane/16 -> K in {16g..16g+15}
// ---------------------------------------------------------------------------
__global__ __launch_bounds__(256)
void gemm_bf16_wmma(const __bf16* __restrict__ A, const __bf16* __restrict__ W,
                    float* __restrict__ C, int M, int N, int K) {
  __shared__ __bf16 ldsB[2][32 * KC];          // 2 x 32KB double buffer

  const int lane = threadIdx.x & 31;
  const int wave = threadIdx.x >> 5;
  const int Mb = M >> 8;                       // 256-row blocks along M
  const int bm = blockIdx.x % Mb;
  const int bn = blockIdx.x / Mb;
  const long m0 = (long)bm * 256 + wave * 32;
  const long n0 = (long)bn * 32;

  const int mr = lane & 15;
  const int g  = lane >> 4;

  v8f acc00 = {}; v8f acc01 = {}; v8f acc10 = {}; v8f acc11 = {};

  const __bf16* aRow0 = A + (m0 + mr) * K;
  const __bf16* aRow1 = A + (m0 + 16 + mr) * K;

  const int nChunks = K / KC;

#if HAVE_TDM
  if (wave == 0)
    tdm_load_strip(W + n0 * K, (unsigned)(uintptr_t)&ldsB[0][0], K);
#endif

  for (int c = 0; c < nChunks; ++c) {
    const int kc  = c * KC;
    const int cur = c & 1;

#if HAVE_TDM
    if (wave == 0) {
      if (c + 1 < nChunks) {
        tdm_load_strip(W + n0 * K + (c + 1) * KC,
                       (unsigned)(uintptr_t)&ldsB[(c + 1) & 1][0], K);
        __builtin_amdgcn_s_wait_tensorcnt(1);  // chunk c's DMA done (in-order)
      } else {
        __builtin_amdgcn_s_wait_tensorcnt(0);
      }
    }
    // The TDM wrote LDS behind the compiler's back: escape the buffer address
    // through a memory-clobbering asm so the ds_loads below cannot be folded.
    {
      unsigned esc = (unsigned)(uintptr_t)&ldsB[0][0];
      asm volatile("" : "+v"(esc)::"memory");
    }
#else
    // Fallback (host pass / toolchain without the TDM builtin):
    // cooperative global->LDS staging.
    {
      const __bf16* src = W + (n0 * K + kc);
      for (int e = threadIdx.x * 8; e < 32 * KC; e += 256 * 8) {
        const int row = e / KC, colk = e % KC;
        *(v8bf*)&ldsB[cur][e] = *(const v8bf*)(src + (long)row * K + colk);
      }
    }
#endif
    __syncthreads();                           // LDS tile visible to all waves

    const __bf16* bbase = &ldsB[cur][0];
    for (int k0 = 0; k0 < KC; k0 += 32) {
      const int ka = kc + k0 + 16 * g;

      v8bf a0lo = *(const v8bf*)(aRow0 + ka);
      v8bf a0hi = *(const v8bf*)(aRow0 + ka + 8);
      v8bf a1lo = *(const v8bf*)(aRow1 + ka);
      v8bf a1hi = *(const v8bf*)(aRow1 + ka + 8);

      v16bf b0 = *(const v16bf*)(bbase + mr * KC + k0 + 16 * g);
      v16bf b1 = *(const v16bf*)(bbase + (16 + mr) * KC + k0 + 16 * g);

      if (k0 + 256 < KC)
        __builtin_prefetch(aRow0 + ka + 256, 0, 1);

      v16bf a0 = __builtin_shufflevector(a0lo, a0hi,
                   0,1,2,3,4,5,6,7,8,9,10,11,12,13,14,15);
      v16bf a1 = __builtin_shufflevector(a1lo, a1hi,
                   0,1,2,3,4,5,6,7,8,9,10,11,12,13,14,15);

      acc00 = __builtin_amdgcn_wmma_f32_16x16x32_bf16(false, a0, false, b0,
                                                      (short)0, acc00, false, false);
      acc01 = __builtin_amdgcn_wmma_f32_16x16x32_bf16(false, a0, false, b1,
                                                      (short)0, acc01, false, false);
      acc10 = __builtin_amdgcn_wmma_f32_16x16x32_bf16(false, a1, false, b0,
                                                      (short)0, acc10, false, false);
      acc11 = __builtin_amdgcn_wmma_f32_16x16x32_bf16(false, a1, false, b1,
                                                      (short)0, acc11, false, false);
    }
    __syncthreads();                           // buffer consumed before reuse
  }

  // D layout: VGPR r, lanes 0-15 -> M=r, N=lane; lanes 16-31 -> M=8+r, N=lane-16
  const int rbase = g * 8;
  const int col   = mr;
#pragma unroll
  for (int r = 0; r < 8; ++r) {
    const long row = m0 + rbase + r;
    C[row * N + n0 + col]             = acc00[r];
    C[row * N + n0 + col + 16]        = acc01[r];
    C[(row + 16) * N + n0 + col]      = acc10[r];
    C[(row + 16) * N + n0 + col + 16] = acc11[r];
  }
}

// ---------------------------------------------------------------------------
// Per-token: depthwise causal conv(4) over x0 channels + SiLU + RMSNorm,
// plus per-head alpha = exp(-exp(A_log)*softplus(dt + dt_bias)).
// ---------------------------------------------------------------------------
__global__ __launch_bounds__(256)
void conv_silu_rms_alpha(const float* __restrict__ zxdt,
                         const float* __restrict__ conv_w,
                         const float* __restrict__ conv_b,
                         const float* __restrict__ norm_w,
                         const float* __restrict__ dt_bias,
                         const float* __restrict__ A_log,
                         float* __restrict__ ssm_x,
                         float* __restrict__ alpha) {
  const int idx = blockIdx.x;            // token = b*L + l
  const int l   = idx & (SEQ_L - 1);
  const int tid = threadIdx.x;
  __shared__ float red[256];

  if (tid < N_HEADS) {
    float dtv = zxdt[(long)idx * E_OUT + 2 * D_INNER + tid] + dt_bias[tid];
    float dt  = (dtv > 20.0f) ? dtv : log1pf(expf(dtv));
    alpha[(long)idx * N_HEADS + tid] = expf(-expf(A_log[tid]) * dt);
  }

  float v[8];
  float ss = 0.0f;
#pragma unroll
  for (int j = 0; j < 8; ++j) {
    const int c = tid + j * 256;
    float acc = conv_b[c];
#pragma unroll
    for (int t = 0; t < D_CONV; ++t) {
      const int lt = l - (D_CONV - 1) + t;
      if (lt >= 0)
        acc += conv_w[c * D_CONV + t] *
               zxdt[(long)(idx - (D_CONV - 1) + t) * E_OUT + D_INNER + c];
    }
    const float s = acc / (1.0f + expf(-acc));  // SiLU
    v[j] = s;
    ss += s * s;
  }
  red[tid] = ss;
  __syncthreads();
  for (int off = 128; off > 0; off >>= 1) {
    if (tid < off) red[tid] += red[tid + off];
    __syncthreads();
  }
  const float rstd = rsqrtf(red[0] / (float)D_INNER + EPSV);
#pragma unroll
  for (int j = 0; j < 8; ++j) {
    const int c = tid + j * 256;
    ssm_x[(long)idx * D_INNER + c] = v[j] * rstd * norm_w[c];
  }
}

// ---------------------------------------------------------------------------
// Chunked linear-recurrence scan, 3 passes.
// s_l = a_l * s_{l-1} + (1-a_l)*x_l ; y_l = s_l + D_h * x_l
// ---------------------------------------------------------------------------
__global__ __launch_bounds__(256)
void scan_pass1(const float* __restrict__ ssm_x, const float* __restrict__ alpha,
                float* __restrict__ chA, float* __restrict__ chB) {
  const int t = blockIdx.x * 256 + threadIdx.x;       // < NCHAIN*NCHUNK
  const int d = t & (HEADDIM - 1);
  const int h = (t >> 6) & (N_HEADS - 1);
  const int c = (t >> 11) & (NCHUNK - 1);
  const int b = t >> 16;
  float pA = 1.0f, pB = 0.0f;
  const long base = (long)b * SEQ_L;
  for (int l = c * CHUNKLEN; l < (c + 1) * CHUNKLEN; ++l) {
    const long row = base + l;
    const float a = alpha[row * N_HEADS + h];
    const float x = ssm_x[row * D_INNER + h * HEADDIM + d];
    pA *= a;
    pB = a * pB + (1.0f - a) * x;
  }
  const long chain = ((long)b * N_HEADS + h) * HEADDIM + d;
  chA[chain * NCHUNK + c] = pA;
  chB[chain * NCHUNK + c] = pB;
}

__global__ void scan_pass2(const float* __restrict__ chA,
                           const float* __restrict__ chB,
                           float* __restrict__ carry) {
  const int chain = blockIdx.x * blockDim.x + threadIdx.x;
  if (chain >= NCHAIN) return;
  float s = 0.0f;
#pragma unroll
  for (int c = 0; c < NCHUNK; ++c) {
    carry[(long)chain * NCHUNK + c] = s;
    s = chA[(long)chain * NCHUNK + c] * s + chB[(long)chain * NCHUNK + c];
  }
}

__global__ __launch_bounds__(256)
void scan_pass3(const float* __restrict__ ssm_x, const float* __restrict__ alpha,
                const float* __restrict__ zxdt, const float* __restrict__ Dp,
                const float* __restrict__ carry, __bf16* __restrict__ ybf) {
  const int t = blockIdx.x * 256 + threadIdx.x;
  const int d = t & (HEADDIM - 1);
  const int h = (t >> 6) & (N_HEADS - 1);
  const int c = (t >> 11) & (NCHUNK - 1);
  const int b = t >> 16;
  const long chain = ((long)b * N_HEADS + h) * HEADDIM + d;
  float s = carry[chain * NCHUNK + c];
  const float Dh = Dp[h];
  const long base = (long)b * SEQ_L;
  for (int l = c * CHUNKLEN; l < (c + 1) * CHUNKLEN; ++l) {
    const long row = base + l;
    const float a = alpha[row * N_HEADS + h];
    const float x = ssm_x[row * D_INNER + h * HEADDIM + d];
    s = a * s + (1.0f - a) * x;
    const float y = s + Dh * x;
    const float z = zxdt[row * E_OUT + h * HEADDIM + d];  // z0 slice
    const float yg = y / (1.0f + expf(-z));               // y * sigmoid(z0)
    ybf[row * D_INNER + h * HEADDIM + d] = (__bf16)yg;
  }
}

// ---------------------------------------------------------------------------
// Launch
// ---------------------------------------------------------------------------
extern "C" void kernel_launch(void* const* d_in, const int* in_sizes, int n_in,
                              void* d_out, int out_size, void* d_ws, size_t ws_size,
                              hipStream_t stream) {
  (void)in_sizes; (void)n_in; (void)out_size; (void)ws_size;
  const float* u       = (const float*)d_in[0];
  const float* W_in    = (const float*)d_in[1];
  // d_in[2]=W_B, d_in[3]=W_C: provably dead (conv channels >= D_INNER unused)
  const float* conv_w  = (const float*)d_in[4];
  const float* conv_b  = (const float*)d_in[5];
  const float* dt_bias = (const float*)d_in[6];
  const float* A_log   = (const float*)d_in[7];
  const float* Dp      = (const float*)d_in[8];
  const float* norm_w  = (const float*)d_in[9];
  const float* W_out   = (const float*)d_in[10];

  char* ws = (char*)d_ws;
  size_t off = 0;
  auto alloc = [&](size_t bytes) -> void* {
    void* p = ws + off;
    off = (off + bytes + 255) & ~(size_t)255;
    return p;
  };
  __bf16* u_bf    = (__bf16*)alloc((size_t)NROWS * D_MODEL * sizeof(__bf16));
  __bf16* Win_bf  = (__bf16*)alloc((size_t)E_OUT * D_MODEL * sizeof(__bf16));
  __bf16* Wout_bf = (__bf16*)alloc((size_t)D_MODEL * D_INNER * sizeof(__bf16));
  float*  zxdt    = (float*)alloc((size_t)NROWS * E_OUT * sizeof(float));
  float*  ssm_x   = (float*)alloc((size_t)NROWS * D_INNER * sizeof(float));
  float*  alpha   = (float*)alloc((size_t)NROWS * N_HEADS * sizeof(float));
  float*  chA     = (float*)alloc((size_t)NCHAIN * NCHUNK * sizeof(float));
  float*  chB     = (float*)alloc((size_t)NCHAIN * NCHUNK * sizeof(float));
  float*  carry   = (float*)alloc((size_t)NCHAIN * NCHUNK * sizeof(float));
  __bf16* ybf     = (__bf16*)alloc((size_t)NROWS * D_INNER * sizeof(__bf16));

  {
    long n = (long)NROWS * D_MODEL;
    cvt_f32_bf16<<<(unsigned)((n + 255) / 256), 256, 0, stream>>>(u, u_bf, n);
  }
  {
    long n = (long)E_OUT * D_MODEL;
    cvt_f32_bf16<<<(unsigned)((n + 255) / 256), 256, 0, stream>>>(W_in, Win_bf, n);
  }
  {
    long n = (long)D_MODEL * D_INNER;
    cvt_f32_bf16<<<(unsigned)((n + 255) / 256), 256, 0, stream>>>(W_out, Wout_bf, n);
  }

  // GEMM1: zxdt[8192,4128] = u_bf[8192,1024] * Win_bf[4128,1024]^T
  {
    const int M = NROWS, N = E_OUT, K = D_MODEL;
    const unsigned blocks = (unsigned)((M / 256) * (N / 32));   // 32 * 129
    gemm_bf16_wmma<<<blocks, 256, 0, stream>>>(u_bf, Win_bf, zxdt, M, N, K);
  }

  conv_silu_rms_alpha<<<NROWS, 256, 0, stream>>>(zxdt, conv_w, conv_b, norm_w,
                                                 dt_bias, A_log, ssm_x, alpha);

  scan_pass1<<<(NCHAIN * NCHUNK) / 256, 256, 0, stream>>>(ssm_x, alpha, chA, chB);
  scan_pass2<<<(NCHAIN + 255) / 256, 256, 0, stream>>>(chA, chB, carry);
  scan_pass3<<<(NCHAIN * NCHUNK) / 256, 256, 0, stream>>>(ssm_x, alpha, zxdt, Dp,
                                                          carry, ybf);

  // GEMM2: out[8192,1024] = ybf[8192,2048] * Wout_bf[1024,2048]^T
  {
    const int M = NROWS, N = D_MODEL, K = D_INNER;
    const unsigned blocks = (unsigned)((M / 256) * (N / 32));   // 32 * 32
    gemm_bf16_wmma<<<blocks, 256, 0, stream>>>(ybf, Wout_bf, (float*)d_out,
                                               M, N, K);
  }
}
